// RavenKnownPrediction_51153060495813
// MI455X (gfx1250) — matrix operand order repaired and checked by reading, not compiled, not run on hardware
//
#include <hip/hip_runtime.h>

typedef __attribute__((ext_vector_type(16))) __bf16 v16bf;
typedef __attribute__((ext_vector_type(8)))  __bf16 v8bf;
typedef __attribute__((ext_vector_type(8)))  float  v8f;
typedef __attribute__((ext_vector_type(4)))  float  f32x4;

#define ZDIM 512
#define NCOL 52
#define NPAD 64
#define KPAD 520          // 512 + 8 bf16 pad -> row stride 1040B (16B aligned, bank-conflict friendly)
#define BM   128          // rows per block (8 waves x 16-row WMMA tiles)

__global__ __launch_bounds__(256)
void raven_fused_gemm_softmax(const float* __restrict__ z,
                              const float* __restrict__ xq,
                              const float* __restrict__ Wg,
                              const float* __restrict__ bias,
                              float* __restrict__ out)
{
    extern __shared__ char smem[];
    __bf16* wlds = (__bf16*)smem;                                   // [NPAD][KPAD] transposed W
    float*  clds = (float*)(smem + (size_t)NPAD * KPAD * sizeof(__bf16)); // [BM][NPAD] logits

    const int tid = threadIdx.x;

    // ---- Stage W (512x52 f32, row-major) into LDS as bf16 transposed [n][k], zero-pad n in [52,64)
    for (int e = tid; e < ZDIM * NCOL; e += 256) {
        int k = e / NCOL;
        int n = e - k * NCOL;
        wlds[n * KPAD + k] = (__bf16)Wg[e];
    }
    for (int e = tid; e < (NPAD - NCOL) * ZDIM; e += 256) {
        int n = NCOL + (e >> 9);
        int k = e & (ZDIM - 1);
        wlds[n * KPAD + k] = (__bf16)0.0f;
    }
    __syncthreads();

    const int wave = tid >> 5;          // wave32
    const int lane = tid & 31;
    const int l16  = lane & 15;
    const int half = lane >> 4;

    const long rowBase = (long)blockIdx.x * BM + wave * 16;
    const float* zrow  = z + (rowBase + l16) * (long)ZDIM;

    v8f acc[4] = {v8f{}, v8f{}, v8f{}, v8f{}};

    for (int ks = 0; ks < ZDIM / 32; ++ks) {
        const int kb = ks * 32 + half * 8;

        // A fragment: 16-bit A 16x32 layout.
        // lanes 0-15 : K[0:8)  in v[0:8),  K[16:24) in v[8:16)
        // lanes 16-31: K[8:16) in v[0:8),  K[24:32) in v[8:16)
        f32x4 f0 = *(const f32x4*)(zrow + kb);
        f32x4 f1 = *(const f32x4*)(zrow + kb + 4);
        f32x4 f2 = *(const f32x4*)(zrow + kb + 16);
        f32x4 f3 = *(const f32x4*)(zrow + kb + 20);

        if (ks + 2 < ZDIM / 32)
            __builtin_prefetch(zrow + kb + 64, 0, 0);   // -> global_prefetch_b8, 2 K-steps ahead

        v16bf afrag;
        afrag[0]  = (__bf16)f0.x; afrag[1]  = (__bf16)f0.y;
        afrag[2]  = (__bf16)f0.z; afrag[3]  = (__bf16)f0.w;
        afrag[4]  = (__bf16)f1.x; afrag[5]  = (__bf16)f1.y;
        afrag[6]  = (__bf16)f1.z; afrag[7]  = (__bf16)f1.w;
        afrag[8]  = (__bf16)f2.x; afrag[9]  = (__bf16)f2.y;
        afrag[10] = (__bf16)f2.z; afrag[11] = (__bf16)f2.w;
        afrag[12] = (__bf16)f3.x; afrag[13] = (__bf16)f3.y;
        afrag[14] = (__bf16)f3.z; afrag[15] = (__bf16)f3.w;

        // B fragments: 16-bit B 32x16 layout; lanes 0-15 hold K[0:8),K[16:24), lanes 16-31 hold K[8:16),K[24:32),
        // lane&15 selects column n. 2x8 bf16 from the transposed LDS image (2x ds_load_b128).
        const int kb2 = ks * 32 + half * 8;
#pragma unroll
        for (int nt = 0; nt < 4; ++nt) {
            const __bf16* wp = wlds + (nt * 16 + l16) * KPAD + kb2;
            v8bf blo = *(const v8bf*)(wp);
            v8bf bhi = *(const v8bf*)(wp + 16);
            v16bf bfrag = __builtin_shufflevector(blo, bhi,
                0, 1, 2, 3, 4, 5, 6, 7, 8, 9, 10, 11, 12, 13, 14, 15);
            acc[nt] = __builtin_amdgcn_wmma_f32_16x16x32_bf16(
                false, afrag, false, bfrag, (short)0, acc[nt], false, false);
        }
    }

    // ---- Scatter accumulators to LDS per C layout: VGPR r -> row r + 8*half, col = nt*16 + (lane&15)
#pragma unroll
    for (int nt = 0; nt < 4; ++nt)
#pragma unroll
        for (int r = 0; r < 8; ++r)
            clds[(wave * 16 + r + half * 8) * NPAD + nt * 16 + l16] = acc[nt][r];

    __syncthreads();

    // ---- Epilogue: one thread per row, 4 groups of 13-way softmax + rule mix
    if (tid < BM) {
        const long gr = (long)blockIdx.x * BM + tid;
        const float* lrow = clds + tid * NPAD;
#pragma unroll
        for (int g = 0; g < 4; ++g) {
            float v[13];
            float mx = -3.0e38f;
#pragma unroll
            for (int j = 0; j < 13; ++j) {
                v[j] = lrow[g * 13 + j] + bias[g * 13 + j];
                mx = fmaxf(mx, v[j]);
            }
            float s = 0.0f;
#pragma unroll
            for (int j = 0; j < 13; ++j) {
                v[j] = __expf(v[j] - mx);
                s += v[j];
            }
            const float aq = xq[gr * 8 + g];
            const float bq = xq[gr * 8 + 4 + g];
            const float rules[13] = {
                aq, bq - 2.0f, bq - 1.0f, bq + 1.0f, bq + 2.0f,
                aq - bq, aq + bq, fminf(aq, bq), fmaxf(aq, bq),
                bq + 2.0f, bq + 1.0f, bq - 2.0f, bq - 1.0f
            };
            float dot = 0.0f;
#pragma unroll
            for (int j = 0; j < 13; ++j)
                dot += rules[j] * v[j];
            out[gr * 4 + g] = dot / s;
        }
    }
}

extern "C" void kernel_launch(void* const* d_in, const int* in_sizes, int n_in,
                              void* d_out, int out_size, void* d_ws, size_t ws_size,
                              hipStream_t stream) {
    const float* z    = (const float*)d_in[0];
    const float* xq   = (const float*)d_in[1];
    const float* Wg   = (const float*)d_in[2];
    const float* bias = (const float*)d_in[3];

    const int B    = in_sizes[0] / ZDIM;     // 262144
    const int grid = B / BM;                 // 2048 blocks
    const size_t shmem = (size_t)NPAD * KPAD * sizeof(__bf16)   // 66,560 B  (W, bf16 transposed)
                       + (size_t)BM * NPAD * sizeof(float);     // 32,768 B  (logit tile)

    // Opt in to >64KB dynamic LDS (WGP has 320KB). Deterministic, capture-safe host call.
    hipFuncSetAttribute(reinterpret_cast<const void*>(raven_fused_gemm_softmax),
                        hipFuncAttributeMaxDynamicSharedMemorySize, (int)shmem);

    raven_fused_gemm_softmax<<<grid, 256, shmem, stream>>>(z, xq, Wg, bias, (float*)d_out);
}
